// FlowDecoder_4501125726393
// MI455X (gfx1250) — compile-verified, matrix operand order checked by reading, not run on hardware
//
#include <hip/hip_runtime.h>

typedef __attribute__((ext_vector_type(16))) __bf16 v16bf;
typedef __attribute__((ext_vector_type(8)))  float  v8f;

// ---------------- geometry constants ----------------
#define NB   8
#define HH   64
#define WW   128
#define HP   66     // padded height
#define WP   130    // padded width
#define CSX  544    // channel stride of activation arena (>= 517, mult of 32)

// arena channel offsets (feat_out order): c5,c4,c3,c2,c1,cost,x1,featup,flowup
// 0,16,48,112,208,336,417,513,515 ; channels 517..543 stay zero.

static constexpr size_t ACT_USHORTS = (size_t)NB * HP * WP * CSX;        // 37,340,160
static constexpr size_t OFF_WF      = ACT_USHORTS * 2;                   // bytes
static constexpr size_t WF_FRAGS    = 2007;                              // total 1KB fragments
static constexpr size_t OFF_FLOWUP  = OFF_WF + WF_FRAGS * 1024;
static constexpr size_t OFF_WARP    = OFF_FLOWUP + (size_t)NB*2*HH*WW*4;
static constexpr size_t WS_NEED     = OFF_WARP + (size_t)NB*96*HH*WW*4;

#if defined(__has_builtin)
#  if __has_builtin(__builtin_amdgcn_global_load_async_to_lds_b128)
#    define HAS_ASYNC 1
#  endif
#endif
#ifndef HAS_ASYNC
#  define HAS_ASYNC 0
#endif

__device__ __forceinline__ unsigned short f2bf(float f) {
    unsigned int u = __float_as_uint(f);
    unsigned int r = u + 0x7fffu + ((u >> 16) & 1u);
    return (unsigned short)(r >> 16);
}
__device__ __forceinline__ float bf2f(unsigned short h) {
    return __uint_as_float(((unsigned int)h) << 16);
}

#if HAS_ASYNC
typedef int v4i_ __attribute__((__vector_size__(4 * sizeof(int))));
typedef __attribute__((address_space(1))) v4i_ as1_v4i;
typedef __attribute__((address_space(3))) v4i_ as3_v4i;
__device__ __forceinline__ void cp16_async(const unsigned short* g, unsigned short* l) {
    __builtin_amdgcn_global_load_async_to_lds_b128(
        (as1_v4i*)const_cast<unsigned short*>(g), (as3_v4i*)l, 0, 0);
}
__device__ __forceinline__ void wait_async0() {
#if __has_builtin(__builtin_amdgcn_s_wait_asynccnt)
    __builtin_amdgcn_s_wait_asynccnt(0);
#else
    asm volatile("s_wait_asynccnt 0" ::: "memory");
#endif
}
#endif

// ---------------- utility kernels ----------------
__global__ void k_zero(uint4* __restrict__ p, int n) {
    int i = blockIdx.x * blockDim.x + threadIdx.x;
    if (i < n) { uint4 z; z.x = z.y = z.z = z.w = 0u; p[i] = z; }
}

// x_1 [8,96,64,128] fp32 NCHW -> arena bf16 channels 417..512
__global__ void k_pack_x1(const float* __restrict__ x1, unsigned short* __restrict__ act, int n) {
    int i = blockIdx.x * blockDim.x + threadIdx.x;
    if (i >= n) return;
    int t = i;
    int x = t % WW; t /= WW;
    int y = t % HH; t /= HH;
    int c = t % 96; t /= 96;
    int b = t;
    act[((size_t)((b * HP + y + 1) * WP + x + 1)) * CSX + 417 + c] = f2bf(x1[i]);
}

// pack conv weights [Cout,Cin,3,3] fp32 -> bf16 WMMA B-fragments
// fragment = 32 lanes x 16 bf16; lane<16 holds K{0..7,16..23}, lane>=16 holds K{8..15,24..31}
__global__ void k_pack_w(const float* __restrict__ w, unsigned short* __restrict__ dst,
                         int Cout, int Cin, int Cpad, int nNt, int nEl) {
    int i = blockIdx.x * blockDim.x + threadIdx.x;
    if (i >= nEl) return;
    int j     = i & 15;
    int lane  = (i >> 4) & 31;
    int frag  = i >> 9;
    int ntile = frag % nNt;
    int chunk = frag / nNt;
    int n     = ntile * 16 + (lane & 15);
    int basek = (lane < 16) ? 0 : 8;
    int klocal = ((j >> 3) * 16) + basek + (j & 7);
    int k   = chunk * 32 + klocal;
    int tap = k / Cpad;
    int c   = k % Cpad;
    float v = 0.f;
    if (n < Cout && c < Cin)
        v = w[((size_t)(n * Cin + c) * 3 + tap / 3) * 3 + (tap % 3)];
    dst[i] = f2bf(v);
}

// ConvTranspose2d k=4 s=2 p=1 : feat_flow [8,545,32,64] -> arena ch 513..514
__global__ void k_deconv_feat(const float* __restrict__ in, const float* __restrict__ w,
                              const float* __restrict__ bias, unsigned short* __restrict__ act, int n) {
    int i = blockIdx.x * blockDim.x + threadIdx.x;
    if (i >= n) return;
    int t = i;
    int ox = t % WW; t /= WW;
    int oy = t % HH; t /= HH;
    int co = t % 2;  t /= 2;
    int b  = t;
    float acc = bias[co];
    int k0y = (oy + 1) & 1, k0x = (ox + 1) & 1;
    for (int ky = k0y; ky < 4; ky += 2) {
        int iy = (oy + 1 - ky) >> 1;
        if (iy < 0 || iy >= 32) continue;
        for (int kx = k0x; kx < 4; kx += 2) {
            int ix = (ox + 1 - kx) >> 1;
            if (ix < 0 || ix >= 64) continue;
            const float* ip = in + (size_t)b * 545 * 2048 + iy * 64 + ix;
            const float* wp = w + (size_t)((co * 4 + ky) * 4 + kx);
            for (int ci = 0; ci < 545; ++ci)
                acc += ip[(size_t)ci * 2048] * wp[(size_t)ci * 32];
        }
    }
    act[((size_t)((b * HP + oy + 1) * WP + ox + 1)) * CSX + 513 + co] = f2bf(acc);
}

// flow [8,2,32,64] -> flow_up fp32 ws + arena ch 515..516
__global__ void k_deconv_flow(const float* __restrict__ in, const float* __restrict__ w,
                              const float* __restrict__ bias, unsigned short* __restrict__ act,
                              float* __restrict__ flowup, int n) {
    int i = blockIdx.x * blockDim.x + threadIdx.x;
    if (i >= n) return;
    int t = i;
    int ox = t % WW; t /= WW;
    int oy = t % HH; t /= HH;
    int co = t % 2;  t /= 2;
    int b  = t;
    float acc = bias[co];
    int k0y = (oy + 1) & 1, k0x = (ox + 1) & 1;
    for (int ky = k0y; ky < 4; ky += 2) {
        int iy = (oy + 1 - ky) >> 1;
        if (iy < 0 || iy >= 32) continue;
        for (int kx = k0x; kx < 4; kx += 2) {
            int ix = (ox + 1 - kx) >> 1;
            if (ix < 0 || ix >= 64) continue;
            for (int ci = 0; ci < 2; ++ci)
                acc += in[((size_t)(b * 2 + ci) * 32 + iy) * 64 + ix] *
                       w[((size_t)(ci * 2 + co) * 4 + ky) * 4 + kx];
        }
    }
    flowup[i] = acc;
    act[((size_t)((b * HP + oy + 1) * WP + ox + 1)) * CSX + 515 + co] = f2bf(acc);
}

// warped = backwarp(x_2, flow_up*5)  -> fp32 ws [8,96,64,128]
__global__ void k_backwarp(const float* __restrict__ x2, const float* __restrict__ flowup,
                           float* __restrict__ warped, int n) {
    int i = blockIdx.x * blockDim.x + threadIdx.x;
    if (i >= n) return;
    int x = i & (WW - 1);
    int y = (i >> 7) & (HH - 1);
    int b = i >> 13;
    float fx = flowup[((size_t)(b * 2 + 0) * HH + y) * WW + x] * 5.0f;
    float fy = flowup[((size_t)(b * 2 + 1) * HH + y) * WW + x] * 5.0f;
    float gx = (-1.0f + (2.0f * x + 1.0f) / (float)WW) + fx / ((WW - 1.0f) * 0.5f);
    float gy = (-1.0f + (2.0f * y + 1.0f) / (float)HH) + fy / ((HH - 1.0f) * 0.5f);
    float xp = (gx + 1.0f) * (WW * 0.5f) - 0.5f;
    float yp = (gy + 1.0f) * (HH * 0.5f) - 0.5f;
    float x0f = floorf(xp), y0f = floorf(yp);
    float wx1 = xp - x0f, wx0 = 1.0f - wx1;
    float wy1 = yp - y0f, wy0 = 1.0f - wy1;
    int x0 = (int)x0f, y0 = (int)y0f, x1 = x0 + 1, y1 = y0 + 1;
    bool vx0 = (x0 >= 0) && (x0 < WW), vx1 = (x1 >= 0) && (x1 < WW);
    bool vy0 = (y0 >= 0) && (y0 < HH), vy1 = (y1 >= 0) && (y1 < HH);
    float w00 = wx0 * wy0, w10 = wx1 * wy0, w01 = wx0 * wy1, w11 = wx1 * wy1;
    bool v00 = vx0 && vy0, v10 = vx1 && vy0, v01 = vx0 && vy1, v11 = vx1 && vy1;
    float wsum = (v00 ? w00 : 0.f) + (v10 ? w10 : 0.f) + (v01 ? w01 : 0.f) + (v11 ? w11 : 0.f);
    float mask = (wsum > 0.999f) ? 1.0f : 0.0f;
    int x0c = min(max(x0, 0), WW - 1), x1c = min(max(x1, 0), WW - 1);
    int y0c = min(max(y0, 0), HH - 1), y1c = min(max(y1, 0), HH - 1);
    const float* xb = x2 + (size_t)b * 96 * HH * WW;
    float* wb = warped + (size_t)b * 96 * HH * WW;
    for (int c = 0; c < 96; ++c) {
        const float* xc = xb + (size_t)c * HH * WW;
        float v = 0.f;
        if (v00) v += w00 * xc[y0c * WW + x0c];
        if (v10) v += w10 * xc[y0c * WW + x1c];
        if (v01) v += w01 * xc[y1c * WW + x0c];
        if (v11) v += w11 * xc[y1c * WW + x1c];
        wb[(size_t)c * HH * WW + y * WW + x] = v * mask;
    }
}

// cost_vol = lrelu(mean_c x1 * shift(warped)) -> arena bf16 ch 336..416
__global__ void k_corr(const float* __restrict__ x1, const float* __restrict__ warped,
                       unsigned short* __restrict__ act, int n) {
    int i = blockIdx.x * blockDim.x + threadIdx.x;
    if (i >= n) return;
    int t = i;
    int x = t % WW; t /= WW;
    int y = t % HH; t /= HH;
    int d = t % 81; t /= 81;
    int b = t;
    int dy = d / 9 - 4, dx = d % 9 - 4;
    int yy = y + dy, xx = x + dx;
    float r = 0.f;
    if (yy >= 0 && yy < HH && xx >= 0 && xx < WW) {
        const float* xa = x1 + (size_t)b * 96 * HH * WW + y * WW + x;
        const float* wa = warped + (size_t)b * 96 * HH * WW + yy * WW + xx;
        for (int c = 0; c < 96; ++c)
            r += xa[(size_t)c * HH * WW] * wa[(size_t)c * HH * WW];
        r *= (1.0f / 96.0f);
    }
    r = (r > 0.f) ? r : 0.1f * r;  // leaky relu
    act[((size_t)((b * HP + y + 1) * WP + x + 1)) * CSX + 336 + d] = f2bf(r);
}

// ---------------- WMMA implicit-GEMM conv3x3 ----------------
// block: 4 waves = 4 M-tiles (16 pixels each), one shared N-group of NT*16 couts.
// B fragments staged in LDS (double-buffered) via async global->LDS; A double-buffered in VGPRs.
template <int NT>
__global__ __launch_bounds__(128)
void k_conv(const unsigned short* __restrict__ act, const unsigned short* __restrict__ wf,
            const float* __restrict__ bias, unsigned short* __restrict__ actOut,
            float* __restrict__ fOut, int cinStart, int nCC, int nNt,
            int coutStart, int coutReal, int mode) {
    __shared__ __align__(16) unsigned short smem[2][NT * 512];

    const int tid  = threadIdx.x;
    const int lane = tid & 31;
    const int wv   = tid >> 5;                    // 0..3
    const int nGroups = nNt / NT;
    const int group   = blockIdx.x % nGroups;
    const int mtile   = (blockIdx.x / nGroups) * 4 + wv;
    const int b  = mtile >> 9;                    // 512 tiles / image
    const int rm = mtile & 511;
    const int y  = rm >> 3;
    const int x0 = (rm & 7) << 4;
    const int ln = lane & 15;
    const int basek = (lane >> 4) << 3;
    const int nCh = 9 * nCC;

    union AU { v16bf v; uint4 q[2]; };
    AU Ab0, Ab1;
    v8f acc[NT];
#pragma unroll
    for (int t = 0; t < NT; ++t)
#pragma unroll
        for (int r = 0; r < 8; ++r) acc[t][r] = 0.0f;

    auto aPtr = [&](int dy, int dx, int cc) -> const unsigned short* {
        return act + (size_t)((b * HP + y + 1 + dy) * WP + (x0 + 1 + dx + ln)) * CSX
                   + cinStart + basek + cc * 32;
    };
    auto stageB = [&](int chunkLin, int buf) {   // weights for chunkLin -> smem[buf]
        const unsigned short* wsrc = wf + ((size_t)(chunkLin * nNt + group * NT)) * 512;
#if HAS_ASYNC
        if (wv == 0) {
#pragma unroll
            for (int q = 0; q < 2 * NT; ++q)
                cp16_async(wsrc + q * 256 + lane * 8, &smem[buf][q * 256 + lane * 8]);
        }
#else
        for (int q = tid; q < NT * 64; q += 128)
            *(uint4*)&smem[buf][q * 8] = *(const uint4*)(wsrc + q * 8);
#endif
    };

    // next-chunk (issue) coordinates
    int ccN = 0, tapN = 0, dyN = -1, dxN = -1;
    auto advance = [&]() {
        if (++ccN == nCC) { ccN = 0; ++tapN; if (++dxN > 1) { dxN = -1; ++dyN; } }
    };

    // prologue: chunk 0
    stageB(0, 0);
    {
        const unsigned short* ap = aPtr(-1, -1, 0);
        Ab0.q[0] = *(const uint4*)ap;
        Ab0.q[1] = *(const uint4*)(ap + 16);
    }
    advance();   // -> coordinates of chunk 1

    auto body = [&](AU& Acur, AU& Anxt, unsigned short* sCur, int nxtBuf, int i) {
#if HAS_ASYNC
        if (wv == 0) wait_async0();
#endif
        __syncthreads();
        if (i + 1 < nCh) {
            stageB(i + 1, nxtBuf);
            const unsigned short* ap = aPtr(dyN, dxN, ccN);
            Anxt.q[0] = *(const uint4*)ap;
            Anxt.q[1] = *(const uint4*)(ap + 16);
            advance();
        }
#pragma unroll
        for (int t = 0; t < NT; ++t) {
            union { v16bf v; uint4 q[2]; } Bf;
            const unsigned short* wl = sCur + t * 512 + lane * 16;
            Bf.q[0] = *(const uint4*)wl;
            Bf.q[1] = *(const uint4*)(wl + 8);
            acc[t] = __builtin_amdgcn_wmma_f32_16x16x32_bf16(
                false, Acur.v, false, Bf.v, (short)0, acc[t], false, false);
        }
    };

    for (int i = 0; i < nCh; i += 2) {
        body(Ab0, Ab1, smem[0], 1, i);
        if (i + 1 < nCh)
            body(Ab1, Ab0, smem[1], 0, i + 1);
    }

#pragma unroll
    for (int t = 0; t < NT; ++t) {
        const int nAbs = (group * NT + t) * 16 + ln;   // C/D layout: N = lane%16
        const float bv = (nAbs < coutReal) ? bias[nAbs] : 0.0f;
#pragma unroll
        for (int r = 0; r < 8; ++r) {
            const int mm = r + (lane >> 4) * 8;        // C/D layout: M = r (+8 upper half)
            float v = acc[t][r] + bv;
            if (mode == 0) {
                v = (v > 0.f) ? v : 0.1f * v;
                actOut[(size_t)((b * HP + y + 1) * WP + (x0 + 1 + mm)) * CSX + coutStart + nAbs] = f2bf(v);
            } else if (nAbs < coutReal) {
                fOut[(((size_t)b * coutReal + nAbs) * HH + y) * WW + (x0 + mm)] = v;
            }
        }
    }
}

// feat_out export: arena bf16 ch 0..516 -> fp32 NCHW
__global__ void k_export(const unsigned short* __restrict__ act, float* __restrict__ out, int n) {
    int i = blockIdx.x * blockDim.x + threadIdx.x;
    if (i >= n) return;
    int t = i;
    int x = t % WW;  t /= WW;
    int y = t % HH;  t /= HH;
    int c = t % 517; t /= 517;
    int b = t;
    out[i] = bf2f(act[((size_t)((b * HP + y + 1) * WP + x + 1)) * CSX + c]);
}

static inline int blks(int n) { return (n + 255) / 256; }

extern "C" void kernel_launch(void* const* d_in, const int* in_sizes, int n_in,
                              void* d_out, int out_size, void* d_ws, size_t ws_size,
                              hipStream_t stream) {
    if (ws_size < WS_NEED) return;

    const float* x1       = (const float*)d_in[0];
    const float* x2       = (const float*)d_in[1];
    const float* featflow = (const float*)d_in[2];
    const float* flow     = (const float*)d_in[3];
    const float* w_upfeat = (const float*)d_in[4];
    const float* b_upfeat = (const float*)d_in[5];
    const float* w_upflow = (const float*)d_in[6];
    const float* b_upflow = (const float*)d_in[7];
    const float* Wc[6] = {(const float*)d_in[8],  (const float*)d_in[10], (const float*)d_in[12],
                          (const float*)d_in[14], (const float*)d_in[16], (const float*)d_in[18]};
    const float* Bc[6] = {(const float*)d_in[9],  (const float*)d_in[11], (const float*)d_in[13],
                          (const float*)d_in[15], (const float*)d_in[17], (const float*)d_in[19]};

    char* ws = (char*)d_ws;
    unsigned short* act    = (unsigned short*)ws;
    unsigned short* wf     = (unsigned short*)(ws + OFF_WF);
    float*          flowup = (float*)(ws + OFF_FLOWUP);
    float*          warped = (float*)(ws + OFF_WARP);
    float*          outFeat = (float*)d_out;
    float*          outFlow = outFeat + (size_t)NB * 517 * HH * WW;

    // layer tables
    const int cinStart[6]  = {336, 208, 112, 48, 16, 0};
    const int CinReal[6]   = {181, 309, 405, 469, 501, 517};
    const int Cpad[6]      = {192, 320, 416, 480, 512, 544};
    const int nCC[6]       = {6, 10, 13, 15, 16, 17};
    const int nNt[6]       = {8, 6, 4, 2, 1, 1};
    const int CoutReal[6]  = {128, 96, 64, 32, 16, 2};
    const int coutStart[6] = {208, 112, 48, 16, 0, 0};
    const int wfOff[6]     = {0, 432, 972, 1440, 1710, 1854};

    // 1) zero the activation arena (halo + channel padding must be 0)
    { int n = (int)(ACT_USHORTS / 8);
      k_zero<<<blks(n), 256, 0, stream>>>((uint4*)act, n); }

    // 2) pack weights into WMMA fragment layout
    for (int l = 0; l < 6; ++l) {
        int nEl = 9 * (Cpad[l] / 32) * nNt[l] * 512;
        k_pack_w<<<blks(nEl), 256, 0, stream>>>(Wc[l], wf + (size_t)wfOff[l] * 512,
                                                CoutReal[l], CinReal[l], Cpad[l], nNt[l], nEl);
    }

    // 3) stage x_1 into the arena (ch 417..512)
    { int n = NB * 96 * HH * WW;
      k_pack_x1<<<blks(n), 256, 0, stream>>>(x1, act, n); }

    // 4) deconvs (feat_up -> ch513..514, flow_up -> fp32 + ch515..516)
    { int n = NB * 2 * HH * WW;
      k_deconv_feat<<<blks(n), 256, 0, stream>>>(featflow, w_upfeat, b_upfeat, act, n);
      k_deconv_flow<<<blks(n), 256, 0, stream>>>(flow, w_upflow, b_upflow, act, flowup, n); }

    // 5) backwarp x_2 with 5*flow_up
    { int n = NB * HH * WW;
      k_backwarp<<<blks(n), 256, 0, stream>>>(x2, flowup, warped, n); }

    // 6) correlation -> lrelu -> arena ch 336..416
    { int n = NB * 81 * HH * WW;
      k_corr<<<blks(n), 256, 0, stream>>>(x1, warped, act, n); }

    // 7) dense conv stack (WMMA bf16, LDS-staged weights)
    const int nMB = NB * HH * (WW / 16) / 4;   // 1024 blocks of 4 M-tiles
    k_conv<4><<<nMB * 2, 128, 0, stream>>>(act, wf + (size_t)wfOff[0] * 512, Bc[0], act, nullptr,
                                           cinStart[0], nCC[0], nNt[0], coutStart[0], CoutReal[0], 0);
    k_conv<2><<<nMB * 3, 128, 0, stream>>>(act, wf + (size_t)wfOff[1] * 512, Bc[1], act, nullptr,
                                           cinStart[1], nCC[1], nNt[1], coutStart[1], CoutReal[1], 0);
    k_conv<4><<<nMB * 1, 128, 0, stream>>>(act, wf + (size_t)wfOff[2] * 512, Bc[2], act, nullptr,
                                           cinStart[2], nCC[2], nNt[2], coutStart[2], CoutReal[2], 0);
    k_conv<2><<<nMB * 1, 128, 0, stream>>>(act, wf + (size_t)wfOff[3] * 512, Bc[3], act, nullptr,
                                           cinStart[3], nCC[3], nNt[3], coutStart[3], CoutReal[3], 0);
    k_conv<1><<<nMB * 1, 128, 0, stream>>>(act, wf + (size_t)wfOff[4] * 512, Bc[4], act, nullptr,
                                           cinStart[4], nCC[4], nNt[4], coutStart[4], CoutReal[4], 0);
    // conv6: fp32 NCHW straight to d_out flow section, no relu
    k_conv<1><<<nMB * 1, 128, 0, stream>>>(act, wf + (size_t)wfOff[5] * 512, Bc[5], act, outFlow,
                                           cinStart[5], nCC[5], nNt[5], coutStart[5], CoutReal[5], 1);

    // 8) export feat_out fp32 NCHW
    { int n = NB * 517 * HH * WW;
      k_export<<<blks(n), 256, 0, stream>>>(act, outFeat, n); }
}
